// Attention_71416716198226
// MI455X (gfx1250) — compile-verified, matrix operand order checked by reading, not run on hardware
//
#include <hip/hip_runtime.h>

// ---------------------------------------------------------------------------
// Problem constants (match reference)
// ---------------------------------------------------------------------------
#define B_  2
#define S_  2048
#define DM_ 2048
#define H_  16
#define HK_ 4
#define G_  4
#define D_  128

typedef __attribute__((ext_vector_type(16))) __bf16 v16bf;
typedef __attribute__((ext_vector_type(8)))  float  v8f;
typedef int v4i_ __attribute__((vector_size(16)));   // matches builtin param type

#ifndef __has_builtin
#define __has_builtin(x) 0
#endif

#if __has_builtin(__builtin_amdgcn_global_load_async_to_lds_b128)
#define HAVE_ASYNC_LDS 1
#else
#define HAVE_ASYNC_LDS 0
#endif

union Frag {
  v16bf v;
  unsigned int u[8];
  unsigned short s[16];
};

__device__ inline unsigned short f2bf(float f) {
  unsigned int u = __float_as_uint(f);
  u += 0x7FFFu + ((u >> 16) & 1u);   // round-to-nearest-even
  return (unsigned short)(u >> 16);
}
__device__ inline float bf2f(unsigned short s) {
  return __uint_as_float(((unsigned int)s) << 16);
}
__device__ inline unsigned int pack2bf(float lo, float hi) {
#if __has_builtin(__builtin_amdgcn_cvt_pk_bf16_f32)
  typedef __attribute__((ext_vector_type(2))) __bf16 v2bf;
  union { v2bf v; unsigned int u; } cv;
  cv.v = __builtin_amdgcn_cvt_pk_bf16_f32(lo, hi);
  return cv.u;
#else
  return (unsigned int)f2bf(lo) | ((unsigned int)f2bf(hi) << 16);
#endif
}

// 16-byte global -> LDS copy; async (ASYNCcnt) when the builtin exists.
__device__ inline void async_copy16(const void* g, void* l) {
#if HAVE_ASYNC_LDS
  typedef v4i_ __attribute__((address_space(1)))* gp_t;   // global <4 x i32>*
  typedef v4i_ __attribute__((address_space(3)))* lp_t;   // LDS    <4 x i32>*
  __builtin_amdgcn_global_load_async_to_lds_b128((gp_t)g, (lp_t)l, 0, 0);
#else
  *(uint4*)l = *(const uint4*)g;
#endif
}
__device__ inline void wait_async() {
#if HAVE_ASYNC_LDS
#if __has_builtin(__builtin_amdgcn_s_wait_asynccnt)
  __builtin_amdgcn_s_wait_asynccnt(0);
#else
  asm volatile("s_wait_asynccnt 0x0" ::: "memory");
#endif
#endif
}

// Load a 16-bit A/B-style fragment for one lane from a 32-element bf16 line.
// ISA layout (16-bit 16x32 A-matrix): lanes 0-15 hold K=0..7 (V0..3) and
// K=16..23 (V4..7); lanes 16-31 hold K=8..15 and K=24..31.
__device__ inline v16bf frag_from_line(const unsigned int* p32, int halfid) {
  Frag f;
  const int kb = halfid * 4;           // uint index of first packed pair
#pragma unroll
  for (int j = 0; j < 4; ++j) {
    f.u[j]     = p32[kb + j];          // K = halfid*8 + 2j .. +1
    f.u[4 + j] = p32[kb + 8 + j];      // K = halfid*8 + 16 + 2j .. +1
  }
  return f.v;
}

__device__ inline v8f wmma_bf16(v16bf a, v16bf b, v8f c) {
  return __builtin_amdgcn_wmma_f32_16x16x32_bf16(
      /*neg_a=*/false, a, /*neg_b=*/false, b,
      /*c_mod=*/(short)0, c, /*reuse_a=*/false, /*reuse_b=*/false);
}

// ---------------------------------------------------------------------------
// f32 -> bf16 elementwise convert (2 elements / thread, packed dword stores)
// ---------------------------------------------------------------------------
__global__ __launch_bounds__(256) void cvt_f32_bf16(const float* __restrict__ in,
                                                    unsigned short* __restrict__ out,
                                                    int n2) {
  int i = blockIdx.x * blockDim.x + threadIdx.x;
  if (i >= n2) return;
  ((unsigned int*)out)[i] = pack2bf(in[2 * i], in[2 * i + 1]);
}

// ---------------------------------------------------------------------------
// W[K][N] f32 -> Wt[N][K] bf16, LDS-tiled 64x64 (coalesced both sides)
// ---------------------------------------------------------------------------
__global__ __launch_bounds__(256) void transpose_cvt(const float* __restrict__ W,
                                                     unsigned short* __restrict__ Wt,
                                                     int K, int N) {
  __shared__ unsigned short t[64 * 66];   // [n][k], padded stride
  const int tid = threadIdx.x;
  const int k0 = blockIdx.y * 64, n0 = blockIdx.x * 64;
#pragma unroll
  for (int i = 0; i < 16; ++i) {
    int e = tid + i * 256;                // 4096 elements
    int r = e >> 6, cc = e & 63;          // r = k row, cc = n col
    t[cc * 66 + r] = f2bf(W[(size_t)(k0 + r) * N + n0 + cc]);
  }
  __syncthreads();
#pragma unroll
  for (int i = 0; i < 8; ++i) {
    int p = tid + i * 256;                // 2048 dwords
    int n = p >> 5, kp = p & 31;
    unsigned int val = (unsigned int)t[n * 66 + 2 * kp] |
                       ((unsigned int)t[n * 66 + 2 * kp + 1] << 16);
    ((unsigned int*)Wt)[((size_t)(n0 + n) * K + k0) / 2 + kp] = val;
  }
}

// ---------------------------------------------------------------------------
// All-BF16 WMMA GEMM: C[MxN] = A[MxK] * Bt[NxK]^T
// A bf16 row-major, Bt bf16 pre-transposed [N][K]. C f32 or bf16 row-major.
// Block = 256 threads (8 waves), tile 64(M) x 128(N), K-step 32.
// Tiles staged via async global->LDS b128 copies (no conversion in loop).
// ---------------------------------------------------------------------------
template <bool OUT_BF16>
__global__ __launch_bounds__(256) void gemm_bf16_wmma(
    const unsigned short* __restrict__ A, const unsigned short* __restrict__ Bt,
    void* __restrict__ Cptr, int M, int N, int K) {
  __shared__ unsigned short ldsA[64 * 32];    // 4 KB
  __shared__ unsigned short ldsB[128 * 32];   // 8 KB  ([n][k])

  const int tid    = threadIdx.x;
  const int wave   = tid >> 5;
  const int lane   = tid & 31;
  const int halfid = lane >> 4;
  const int c      = lane & 15;
  const int mb     = halfid * 8;

  const int m0   = blockIdx.y * 64;
  const int n0   = blockIdx.x * 128;
  const int moff = (wave & 1) * 32;
  const int noff = (wave >> 1) * 32;

  v8f acc[2][2] = {};

  for (int k0 = 0; k0 < K; k0 += 32) {
    __syncthreads();
    // A tile: 64 rows x 32 bf16 = 256 x 16B chunks, 1 per thread
    {
      int row = tid >> 2, cp = tid & 3;
      async_copy16(&A[(size_t)(m0 + row) * K + k0 + cp * 8],
                   &ldsA[row * 32 + cp * 8]);
    }
    // B tile: 128 n-rows x 32 bf16 = 512 chunks, 2 per thread
#pragma unroll
    for (int i = 0; i < 2; ++i) {
      int p = tid + i * 256;
      int n = p >> 2, cp = p & 3;
      async_copy16(&Bt[(size_t)(n0 + n) * K + k0 + cp * 8],
                   &ldsB[n * 32 + cp * 8]);
    }
    wait_async();
    __syncthreads();

    v16bf aF[2], bF[2];
#pragma unroll
    for (int i = 0; i < 2; ++i)
      aF[i] = frag_from_line((const unsigned int*)&ldsA[(moff + 16 * i + c) * 32], halfid);
#pragma unroll
    for (int j = 0; j < 2; ++j)
      bF[j] = frag_from_line((const unsigned int*)&ldsB[(noff + 16 * j + c) * 32], halfid);
#pragma unroll
    for (int i = 0; i < 2; ++i)
#pragma unroll
      for (int j = 0; j < 2; ++j)
        acc[i][j] = wmma_bf16(aF[i], bF[j], acc[i][j]);
  }

  // epilogue: C/D layout — lane half gives M base, lane%16 gives N column
#pragma unroll
  for (int i = 0; i < 2; ++i)
#pragma unroll
    for (int j = 0; j < 2; ++j)
#pragma unroll
      for (int v = 0; v < 8; ++v) {
        int row = m0 + moff + 16 * i + mb + v;
        int col = n0 + noff + 16 * j + c;
        float val = acc[i][j][v];
        if (OUT_BF16)
          ((unsigned short*)Cptr)[(size_t)row * N + col] = f2bf(val);
        else
          ((float*)Cptr)[(size_t)row * N + col] = val;
      }
}

// ---------------------------------------------------------------------------
// RoPE on bf16 tensor laid out [B, S, heads, D]. One thread per (b,s,h,i<64).
// ---------------------------------------------------------------------------
__global__ __launch_bounds__(256) void rope_kernel(unsigned short* __restrict__ q,
                                                   int heads, int total) {
  int idx = blockIdx.x * blockDim.x + threadIdx.x;
  if (idx >= total) return;
  int i  = idx & 63;
  int h  = (idx >> 6) % heads;
  int bs = idx / (64 * heads);               // b*S + s
  int s  = bs % S_;
  size_t base = ((size_t)bs * heads + h) * D_;
  float inv = __expf(-(float)(2 * i) * (9.210340371976184f / 128.0f));
  float ang = (float)s * inv;
  float cs = __cosf(ang), sn = __sinf(ang);
  float x1 = bf2f(q[base + i]);
  float x2 = bf2f(q[base + i + 64]);
  q[base + i]      = f2bf(x1 * cs - x2 * sn);
  q[base + i + 64] = f2bf(x2 * cs + x1 * sn);
}

// ---------------------------------------------------------------------------
// Causal flash attention, GQA. Block = 128 threads (4 waves), each wave owns
// 16 query rows; 32-key tiles staged in LDS (K via async copies) and shared.
// Q,K,V: bf16 [B,S,heads,D]. O: bf16 [B,S,H,D].
// ---------------------------------------------------------------------------
__global__ __launch_bounds__(128) void flash_attn_kernel(
    const unsigned short* __restrict__ Q, const unsigned short* __restrict__ Kg,
    const unsigned short* __restrict__ Vg, unsigned short* __restrict__ O) {
  __shared__ unsigned short ldsK[32 * 128];   // [key][d]     8 KB
  __shared__ unsigned short ldsVt[128 * 32];  // [d][key]     8 KB
  __shared__ unsigned short ldsP[4 * 16 * 32];

  const int tid    = threadIdx.x;
  const int wave   = tid >> 5;
  const int lane   = tid & 31;
  const int halfid = lane >> 4;
  const int c      = lane & 15;
  const int mb     = halfid * 8;

  const int qt = blockIdx.x;           // 64-row query tile
  const int h  = blockIdx.y;
  const int b  = blockIdx.z;
  const int kh = h >> 2;               // G = 4
  const int q0 = qt * 64 + wave * 16;  // this wave's first query row
  const float scale = 0.0883883476483184f;  // 1/sqrt(128)

  // Q fragments for 16 rows x 128 D (4 K-chunks of 32), lane row = q0 + c
  v16bf qF[4];
  {
    size_t qoff = (((size_t)b * S_ + q0 + c) * H_ + h) * D_;
    const unsigned int* qp = (const unsigned int*)(Q + qoff);
#pragma unroll
    for (int kc = 0; kc < 4; ++kc)
      qF[kc] = frag_from_line(qp + kc * 16, halfid);
  }

  v8f oacc[8] = {};                    // 8 d-subtiles of 16x16
  float mrow[8], lrow[8];
#pragma unroll
  for (int v = 0; v < 8; ++v) { mrow[v] = -3.0e38f; lrow[v] = 0.0f; }

  const int nT = 2 * (qt + 1);         // causal horizon for this block
  for (int t = 0; t < nT; ++t) {
    const int kbase = t * 32;
    __syncthreads();
    // --- K tile via async global->LDS: 32 keys x 256B = 512 chunks ---
    {
      size_t krow = (((size_t)b * S_ + kbase) * HK_ + kh) * D_;
#pragma unroll
      for (int i = 0; i < 4; ++i) {
        int p = tid + i * 128;
        int key = p >> 4, cp = p & 15;
        async_copy16(&Kg[krow + (size_t)key * (HK_ * D_) + cp * 8],
                     &ldsK[key * 128 + cp * 8]);
      }
    }
    // --- V tile, transposed into [d][key] during staging ---
#pragma unroll
    for (int i = 0; i < 16; ++i) {
      int p   = tid + i * 128;          // 2048 dword slots
      int key = p >> 6, dp = p & 63;
      size_t goff = ((((size_t)b * S_ + kbase + key) * HK_ + kh) * D_) / 2 + dp;
      unsigned int vvv = ((const unsigned int*)Vg)[goff];
      ldsVt[(dp * 2) * 32 + key]     = (unsigned short)(vvv & 0xFFFFu);
      ldsVt[(dp * 2 + 1) * 32 + key] = (unsigned short)(vvv >> 16);
    }
    wait_async();
    __syncthreads();

    // --- scores S = Q * K^T : two 16x16 tiles over 32 keys ---
    v8f sacc[2] = {};
#pragma unroll
    for (int n = 0; n < 2; ++n) {
      const unsigned short* kp = &ldsK[(n * 16 + c) * 128];
#pragma unroll
      for (int kc = 0; kc < 4; ++kc)
        sacc[n] = wmma_bf16(qF[kc],
                            frag_from_line((const unsigned int*)(kp + kc * 32), halfid),
                            sacc[n]);
    }

    // --- online softmax (row = q0+mb+v lives in one 16-lane half) ---
    float pp[2][8], tm[8];
#pragma unroll
    for (int v = 0; v < 8; ++v) {
      int qrow = q0 + mb + v;
#pragma unroll
      for (int n = 0; n < 2; ++n) {
        int key = kbase + n * 16 + c;
        float sval = sacc[n][v] * scale;
        pp[n][v] = (key <= qrow) ? sval : -3.0e38f;
      }
      tm[v] = fmaxf(pp[0][v], pp[1][v]);
    }
#pragma unroll
    for (int off = 8; off >= 1; off >>= 1)
#pragma unroll
      for (int v = 0; v < 8; ++v)
        tm[v] = fmaxf(tm[v], __shfl_xor(tm[v], off, 16));

    float corr[8];
#pragma unroll
    for (int v = 0; v < 8; ++v) {
      float nm = fmaxf(mrow[v], tm[v]);
      corr[v]  = __expf(mrow[v] - nm);
      mrow[v]  = nm;
      pp[0][v] = __expf(pp[0][v] - nm);
      pp[1][v] = __expf(pp[1][v] - nm);
    }
    float rs[8];
#pragma unroll
    for (int v = 0; v < 8; ++v) rs[v] = pp[0][v] + pp[1][v];
#pragma unroll
    for (int off = 8; off >= 1; off >>= 1)
#pragma unroll
      for (int v = 0; v < 8; ++v)
        rs[v] += __shfl_xor(rs[v], off, 16);
#pragma unroll
    for (int v = 0; v < 8; ++v) lrow[v] = lrow[v] * corr[v] + rs[v];
#pragma unroll
    for (int d = 0; d < 8; ++d)
#pragma unroll
      for (int v = 0; v < 8; ++v) oacc[d][v] *= corr[v];

    // --- re-fragment P via LDS (C-layout -> A-layout), then O += P*V ---
    unsigned short* pw = &ldsP[wave * 16 * 32];
#pragma unroll
    for (int v = 0; v < 8; ++v) {
      pw[(mb + v) * 32 + c]      = f2bf(pp[0][v]);
      pw[(mb + v) * 32 + 16 + c] = f2bf(pp[1][v]);
    }
    asm volatile("s_wait_dscnt 0x0" ::: "memory");  // wave-local LDS RAW fence
    v16bf pF = frag_from_line((const unsigned int*)&pw[c * 32], halfid);
#pragma unroll
    for (int d = 0; d < 8; ++d) {
      v16bf vF = frag_from_line((const unsigned int*)&ldsVt[(d * 16 + c) * 32], halfid);
      oacc[d] = wmma_bf16(pF, vF, oacc[d]);
    }
  }

  // --- normalize + store bf16 [B,S,H,D] ---
  float inv[8];
#pragma unroll
  for (int v = 0; v < 8; ++v) inv[v] = 1.0f / lrow[v];
#pragma unroll
  for (int d = 0; d < 8; ++d)
#pragma unroll
    for (int v = 0; v < 8; ++v) {
      size_t off = (((size_t)b * S_ + q0 + mb + v) * H_ + h) * D_ + d * 16 + c;
      O[off] = f2bf(oacc[d][v] * inv[v]);
    }
}

// ---------------------------------------------------------------------------
// Pipeline: convert x/weights to bf16 (weights transposed) -> QKV GEMMs ->
// RoPE -> flash attention -> output GEMM.
// Workspace (bf16): xb 16M | Wqt 8M | Wkt 2M | Wvt 2M | Wot 8M |
//                   Q 16M | K 4M | V 4M | attn-out 16M   (~76 MiB)
// ---------------------------------------------------------------------------
extern "C" void kernel_launch(void* const* d_in, const int* in_sizes, int n_in,
                              void* d_out, int out_size, void* d_ws, size_t ws_size,
                              hipStream_t stream) {
  (void)in_sizes; (void)n_in; (void)out_size; (void)ws_size;
  const float* x  = (const float*)d_in[0];
  const float* Wq = (const float*)d_in[1];
  const float* Wk = (const float*)d_in[2];
  const float* Wv = (const float*)d_in[3];
  const float* Wo = (const float*)d_in[4];

  const int M = B_ * S_;  // 4096
  char* ws = (char*)d_ws;
  size_t off = 0;
  auto take = [&](size_t bytes) { char* p = ws + off; off += bytes; return p; };
  unsigned short* xb  = (unsigned short*)take((size_t)M * DM_ * 2);           // 16M
  unsigned short* Wqt = (unsigned short*)take((size_t)(H_ * D_) * DM_ * 2);   //  8M
  unsigned short* Wkt = (unsigned short*)take((size_t)(HK_ * D_) * DM_ * 2);  //  2M
  unsigned short* Wvt = (unsigned short*)take((size_t)(HK_ * D_) * DM_ * 2);  //  2M
  unsigned short* Wot = (unsigned short*)take((size_t)DM_ * (H_ * D_) * 2);   //  8M
  unsigned short* Qb  = (unsigned short*)take((size_t)B_ * S_ * H_  * D_ * 2);
  unsigned short* Kb  = (unsigned short*)take((size_t)B_ * S_ * HK_ * D_ * 2);
  unsigned short* Vb  = (unsigned short*)take((size_t)B_ * S_ * HK_ * D_ * 2);
  unsigned short* Ab  = (unsigned short*)take((size_t)B_ * S_ * H_  * D_ * 2);

  // One-time operand conversion: x -> bf16; weights -> bf16 transposed [N][K]
  cvt_f32_bf16<<<(M * DM_ / 2) / 256, 256, 0, stream>>>(x, xb, M * DM_ / 2);
  transpose_cvt<<<dim3((H_ * D_) / 64, DM_ / 64), 256, 0, stream>>>(Wq, Wqt, DM_, H_ * D_);
  transpose_cvt<<<dim3((HK_ * D_) / 64, DM_ / 64), 256, 0, stream>>>(Wk, Wkt, DM_, HK_ * D_);
  transpose_cvt<<<dim3((HK_ * D_) / 64, DM_ / 64), 256, 0, stream>>>(Wv, Wvt, DM_, HK_ * D_);
  transpose_cvt<<<dim3(DM_ / 64, (H_ * D_) / 64), 256, 0, stream>>>(Wo, Wot, H_ * D_, DM_);

  // Projections (bf16 x bf16 -> bf16)
  gemm_bf16_wmma<true><<<dim3((H_ * D_) / 128, M / 64), 256, 0, stream>>>(
      xb, Wqt, Qb, M, H_ * D_, DM_);
  gemm_bf16_wmma<true><<<dim3((HK_ * D_) / 128, M / 64), 256, 0, stream>>>(
      xb, Wkt, Kb, M, HK_ * D_, DM_);
  gemm_bf16_wmma<true><<<dim3((HK_ * D_) / 128, M / 64), 256, 0, stream>>>(
      xb, Wvt, Vb, M, HK_ * D_, DM_);

  // RoPE on Q and K
  const int qTot = B_ * S_ * H_ * 64;
  rope_kernel<<<qTot / 256, 256, 0, stream>>>(Qb, H_, qTot);
  const int kTot = B_ * S_ * HK_ * 64;
  rope_kernel<<<kTot / 256, 256, 0, stream>>>(Kb, HK_, kTot);

  // Flash attention (causal, GQA)
  flash_attn_kernel<<<dim3(S_ / 64, H_, B_), 128, 0, stream>>>(Qb, Kb, Vb, Ab);

  // Output projection (bf16 x bf16 -> f32 into d_out)
  gemm_bf16_wmma<false><<<dim3(DM_ / 128, M / 64), 256, 0, stream>>>(
      Ab, Wot, (float*)d_out, M, DM_, DM_);
}
